// GraphDecoder_48747878810071
// MI455X (gfx1250) — matrix operand order repaired and checked by reading, not compiled, not run on hardware
//
#include <hip/hip_runtime.h>
#include <math.h>
#include <stdint.h>

#define N_      30000
#define E_      60000
#define B_      256
#define H_      256
#define DE_     256
#define L_      32
#define V_      40
#define NB_     4
#define MAXNB_  6
#define T_      6
#define DEPTH_  4
#define PT_     1536
#define PA_     1280
#define PB_     2560

typedef __attribute__((ext_vector_type(16))) __bf16 v16bf;
typedef __attribute__((ext_vector_type(8)))  __bf16 v8bf;
typedef __attribute__((ext_vector_type(8)))  float  v8f;

__device__ __forceinline__ __bf16 f2bf(float f) {
  unsigned int u = __builtin_bit_cast(unsigned int, f);
  u += 0x7fffu + ((u >> 16) & 1u);               // round-to-nearest-even
  unsigned short s = (unsigned short)(u >> 16);
  return __builtin_bit_cast(__bf16, s);
}

// ---------------- generic WMMA GEMM: C = epilogue(concat_segs(A) @ Wbf16^T) --
struct ASeg {
  const float* base;   // null => one-hot segment
  const int*   rowidx; // null => identity row
  const int*   ohidx;  // one-hot index per row (when base==null)
  int ld;
  int width;
};

struct GemmArgs {
  ASeg seg[4];
  int nseg;
  const __bf16* W;      // [NOUT, ldW] bf16 (K-major, zero-padded to ldW)
  const float* bias;    // [NOUT] or null
  const float* addend;  // [M, NOUT] or null
  const int*   rowmask; // null ok; row kept iff rowmask[row]!=0
  float* C;             // [M, NOUT]
  int M, K, NOUT, ldW;  // ldW = round_up(K, 64)
  int relu;
  int zero_row0;
};

__device__ __forceinline__ float load_a_elem(const GemmArgs& g, int m, int k) {
#pragma unroll
  for (int s = 0; s < 4; ++s) {
    if (s >= g.nseg) break;
    const ASeg sg = g.seg[s];
    if (k < sg.width) {
      if (sg.base) {
        int r = sg.rowidx ? sg.rowidx[m] : m;
        return sg.base[(long long)r * sg.ld + k];
      }
      return (sg.ohidx[m] == k) ? 1.0f : 0.0f;
    }
    k -= sg.width;
  }
  return 0.0f;
}

__global__ __launch_bounds__(256) void wmma_gemm(GemmArgs g) {
  __shared__ __attribute__((aligned(64))) __bf16 As[2][128][64]; // [m][k] row-major
  __shared__ __attribute__((aligned(64))) __bf16 Bs[2][64][64];  // [n][k] K-major

  const int tid  = threadIdx.x;
  const int lane = tid & 31;
  const int wave = tid >> 5;
  const int hf   = lane >> 4;   // which 16-lane half
  const int mrow = lane & 15;

  const int m0 = blockIdx.y * 128;
  const int n0 = blockIdx.x * 64;

  v8f acc[4];
#pragma unroll
  for (int t = 0; t < 4; ++t)
#pragma unroll
    for (int i = 0; i < 8; ++i) acc[t][i] = 0.0f;

  // per-thread constant staging coordinates
  const int a_mm    = tid >> 1;           // A: row within tile
  const int a_kbase = (tid & 1) << 5;     // A: 32 contiguous k
  const int b_nn    = tid >> 2;           // B: column within tile
  const int b_kb    = (tid & 3) << 4;     // B: 16 contiguous k

  auto stage = [&](int kc, int buf) {
    const int k0 = kc << 6;
    { // ---- stage B tile 64x64: async copy memory -> LDS (bf16, final layout)
      const __bf16* wp = g.W + (long long)(n0 + b_nn) * g.ldW + k0 + b_kb;
      unsigned long long ga = (unsigned long long)wp;
      unsigned lds = (unsigned)(uintptr_t)&Bs[buf][b_nn][b_kb];
      asm volatile("global_load_async_to_lds_b128 %0, %1, off"
                   :: "v"(lds), "v"(ga) : "memory");
      asm volatile("global_load_async_to_lds_b128 %0, %1, off offset:16"
                   :: "v"(lds), "v"(ga) : "memory");
    }
    { // ---- stage A tile 128x64 (f32 -> bf16), 32 contiguous k per thread ----
      const int mg     = m0 + a_mm;
      const int kstart = k0 + a_kbase;
      bool fast = false;
      if (mg < g.M) {
        int koff = kstart;
#pragma unroll
        for (int s = 0; s < 4; ++s) {
          if (s >= g.nseg) break;
          const ASeg sg = g.seg[s];
          if (koff < sg.width) {
            if (sg.base && koff + 32 <= sg.width) {
              const int r = sg.rowidx ? sg.rowidx[mg] : mg;
              const float4* p = (const float4*)(sg.base + (long long)r * sg.ld + koff);
#pragma unroll
              for (int q = 0; q < 4; ++q) {          // 8x float4 = 32 floats
                float4 v0 = p[2 * q + 0];
                float4 v1 = p[2 * q + 1];
                v8bf pk;
                pk[0] = f2bf(v0.x); pk[1] = f2bf(v0.y);
                pk[2] = f2bf(v0.z); pk[3] = f2bf(v0.w);
                pk[4] = f2bf(v1.x); pk[5] = f2bf(v1.y);
                pk[6] = f2bf(v1.z); pk[7] = f2bf(v1.w);
                *(v8bf*)&As[buf][a_mm][a_kbase + 8 * q] = pk;
              }
              fast = true;
            }
            break;
          }
          koff -= sg.width;
        }
      }
      if (!fast) { // boundary-crossing / out-of-range: scalar fallback
        for (int j = 0; j < 32; ++j) {
          const int kg = kstart + j;
          float v = 0.0f;
          if (mg < g.M && kg < g.K) v = load_a_elem(g, mg, kg);
          As[buf][a_mm][a_kbase + j] = f2bf(v);
        }
      }
    }
  };

  auto compute = [&](int buf) {
#pragma unroll
    for (int kk2 = 0; kk2 < 64; kk2 += 32) {
      // A fragment: lane hf/mrow -> K = hf*16 .. +15 (contiguous)
      v16bf afrag = *(const v16bf*)&As[buf][wave * 16 + mrow][kk2 + hf * 16];
#pragma unroll
      for (int nt = 0; nt < 4; ++nt) {
        // B fragment: column n = nt*16+mrow, K = kk2 + hf*16 .. +15 (contiguous)
        v16bf bfrag = *(const v16bf*)&Bs[buf][nt * 16 + mrow][kk2 + hf * 16];
        acc[nt] = __builtin_amdgcn_wmma_f32_16x16x32_bf16(
            false, afrag, false, bfrag, (short)0, acc[nt], false, false);
      }
    }
  };

  const int nchunks = g.ldW >> 6;
  // software pipeline with LDS double buffering: one barrier per K-chunk
  stage(0, 0);
  asm volatile("s_wait_asynccnt 0x0" ::: "memory");
  __syncthreads();
  for (int kc = 0; kc < nchunks; ++kc) {
    const int cur = kc & 1;
    if (kc + 1 < nchunks) stage(kc + 1, cur ^ 1);
    compute(cur);
    asm volatile("s_wait_asynccnt 0x0" ::: "memory");
    __syncthreads();
  }

  // epilogue: VGPR r -> M = r + hf*8 within tile, N = mrow
#pragma unroll
  for (int nt = 0; nt < 4; ++nt) {
#pragma unroll
    for (int r = 0; r < 8; ++r) {
      const int row = m0 + wave * 16 + hf * 8 + r;
      const int col = n0 + nt * 16 + mrow;
      if (row < g.M) {
        float v = acc[nt][r];
        if (g.bias)   v += g.bias[col];
        if (g.addend) v += g.addend[(long long)row * g.NOUT + col];
        if (g.relu)   v  = fmaxf(v, 0.0f);
        if (g.zero_row0 && row == 0) v = 0.0f;
        if (g.rowmask && g.rowmask[row] == 0) v = 0.0f;
        g.C[(long long)row * g.NOUT + col] = v;
      }
    }
  }
}

// ---------------- helper kernels ----------------
__global__ void zero_f32_kernel(float* p, long long n) {
  long long t = (long long)blockIdx.x * blockDim.x + threadIdx.x;
  if (t < n) p[t] = 0.0f;
}

// W [K, NOUT] f32 -> Wt [NOUT, ldW] bf16, zero-padded
__global__ void cvt_transpose_bf16(const float* __restrict__ in, __bf16* __restrict__ out,
                                   int K, int NOUT, int ldW) {
  long long t = (long long)blockIdx.x * blockDim.x + threadIdx.x;
  long long total = (long long)NOUT * ldW;
  if (t >= total) return;
  int n = (int)(t / ldW), k = (int)(t % ldW);
  out[t] = (k < K) ? f2bf(in[(long long)k * NOUT + n]) : f2bf(0.0f);
}

__global__ void mask_graph_kernel(const int* __restrict__ g, const int* __restrict__ em,
                                  int* __restrict__ out, long long n) {
  long long t = (long long)blockIdx.x * blockDim.x + threadIdx.x;
  if (t >= n) return;
  int mi = g[t];
  out[t] = mi * em[mi];
}

__global__ void combine_idx_kernel(const int* __restrict__ a, const int* __restrict__ b,
                                   int scale, int* __restrict__ out, int n) {
  int t = blockIdx.x * blockDim.x + threadIdx.x;
  if (t < n) out[t] = a[t] * scale + b[t];
}

__global__ void init_h_kernel(const float* __restrict__ pre, const float* __restrict__ bh,
                              float* __restrict__ h) {
  long long t = (long long)blockIdx.x * blockDim.x + threadIdx.x;
  if (t >= (long long)E_ * H_) return;
  int e = (int)(t / H_), k = (int)(t % H_);
  float v = fmaxf(pre[t] + bh[k], 0.0f);
  h[t] = (e == 0) ? 0.0f : v;
}

__global__ void gather_sum6_kernel(const float* __restrict__ h, const int* __restrict__ idx,
                                   float* __restrict__ out, int M) {
  const int chunks = H_ / 4;
  long long t = (long long)blockIdx.x * blockDim.x + threadIdx.x;
  if (t >= (long long)M * chunks) return;
  int m = (int)(t / chunks), c = (int)(t % chunks);
  const int* row = idx + (long long)m * MAXNB_;
  float4 s = make_float4(0.f, 0.f, 0.f, 0.f);
#pragma unroll
  for (int j = 0; j < MAXNB_; ++j) {
    const float4 v = *(const float4*)(h + (long long)row[j] * H_ + c * 4);
    s.x += v.x; s.y += v.y; s.z += v.z; s.w += v.w;
  }
  *(float4*)(out + (long long)m * H_ + c * 4) = s;
}

__global__ void segsum_kernel(const float* __restrict__ nodes_t, const int* __restrict__ n2g,
                              float* __restrict__ gvec_t) {
  long long t = (long long)blockIdx.x * blockDim.x + threadIdx.x;
  if (t >= (long long)N_ * H_) return;
  int n = (int)(t / H_), k = (int)(t % H_);
  atomicAdd(&gvec_t[(long long)n2g[n] * H_ + k], nodes_t[t]);
}

__global__ void hist_kernel(const float* __restrict__ wb, const int* __restrict__ gs,
                            float* __restrict__ hist) {
  long long t = (long long)blockIdx.x * blockDim.x + threadIdx.x;
  if (t >= (long long)PB_ * H_) return;
  int m = (int)(t / H_), k = (int)(t % H_);
  float s = 0.0f;
  for (int i = gs[m]; i < m; ++i) s += wb[(long long)i * H_ + k];
  hist[t] = s;
}

__global__ void topo_loss_kernel(const float* __restrict__ hid, const float* __restrict__ w2,
                                 const float* __restrict__ b2, const int* __restrict__ label,
                                 float* out, float inv_b) {
  int wid  = (int)(((long long)blockIdx.x * blockDim.x + threadIdx.x) >> 5);
  int lane = threadIdx.x & 31;
  if (wid >= PT_) return;
  float s = 0.0f;
  for (int k = lane; k < H_; k += 32) s += hid[(long long)wid * H_ + k] * w2[k];
#pragma unroll
  for (int off = 16; off; off >>= 1) s += __shfl_xor(s, off, 32);
  if (lane == 0) {
    float tl = s + b2[0];
    float sp = fmaxf(tl, 0.0f) + log1pf(expf(-fabsf(tl)));
    atomicAdd(out, (sp - tl * (float)label[wid]) * inv_b);
  }
}

__global__ void atom_loss_kernel(const float* __restrict__ hid, const float* __restrict__ w2,
                                 const float* __restrict__ b2, const int* __restrict__ label,
                                 float* out, float inv_b) {
  __shared__ float logits[64];
  int m = blockIdx.x;
  int v = threadIdx.x;
  if (v < V_) {
    float s = b2[v];
    for (int k = 0; k < H_; ++k) s += hid[(long long)m * H_ + k] * w2[k * V_ + v];
    logits[v] = s;
  }
  __syncthreads();
  if (v == 0) {
    float mx = logits[0];
    for (int i = 1; i < V_; ++i) mx = fmaxf(mx, logits[i]);
    float se = 0.0f;
    for (int i = 0; i < V_; ++i) se += expf(logits[i] - mx);
    float lse = mx + logf(se);
    atomicAdd(out, (lse - logits[label[m]]) * inv_b);
  }
}

__global__ void bond_loss_kernel(const float* __restrict__ hid, const float* __restrict__ w2,
                                 const float* __restrict__ b2, const int* __restrict__ label,
                                 float* out, float inv_b) {
  int m = blockIdx.x * blockDim.x + threadIdx.x;
  if (m >= PB_) return;
  float lg[NB_];
#pragma unroll
  for (int j = 0; j < NB_; ++j) lg[j] = b2[j];
  for (int k = 0; k < H_; ++k) {
    float hv = hid[(long long)m * H_ + k];
#pragma unroll
    for (int j = 0; j < NB_; ++j) lg[j] += hv * w2[k * NB_ + j];
  }
  float mx = lg[0];
#pragma unroll
  for (int j = 1; j < NB_; ++j) mx = fmaxf(mx, lg[j]);
  float se = 0.0f;
#pragma unroll
  for (int j = 0; j < NB_; ++j) se += expf(lg[j] - mx);
  float lse = mx + logf(se);
  atomicAdd(out, (lse - lg[label[m]]) * inv_b);
}

// ---------------- host ----------------
static inline ASeg segf(const float* base, const int* rows, int ld, int width) {
  ASeg s; s.base = base; s.rowidx = rows; s.ohidx = nullptr; s.ld = ld; s.width = width; return s;
}
static inline ASeg segoh(const int* oh, int width) {
  ASeg s; s.base = nullptr; s.rowidx = nullptr; s.ohidx = oh; s.ld = 0; s.width = width; return s;
}
static inline dim3 grid1(long long n) { return dim3((unsigned)((n + 255) / 256)); }
static inline int rup64(int k) { return (k + 63) & ~63; }

static void launch_gemm(const GemmArgs& g, hipStream_t s) {
  dim3 grid(g.NOUT / 64, (unsigned)((g.M + 127) / 128));
  wmma_gemm<<<grid, 256, 0, s>>>(g);
}

extern "C" void kernel_launch(void* const* d_in, const int* in_sizes, int n_in,
                              void* d_out, int out_size, void* d_ws, size_t ws_size,
                              hipStream_t stream) {
  const float* src      = (const float*)d_in[0];
  const float* fnode    = (const float*)d_in[1];
  const float* fmess    = (const float*)d_in[2];
  const float* W_h      = (const float*)d_in[3];
  const float* U_h      = (const float*)d_in[4];
  const float* b_h      = (const float*)d_in[5];
  const float* W_o      = (const float*)d_in[6];
  const float* b_o      = (const float*)d_in[7];
  const float* topo_w1  = (const float*)d_in[8];
  const float* topo_b1  = (const float*)d_in[9];
  const float* topo_w2  = (const float*)d_in[10];
  const float* topo_b2  = (const float*)d_in[11];
  const float* atom_w1  = (const float*)d_in[12];
  const float* atom_b1  = (const float*)d_in[13];
  const float* atom_w2  = (const float*)d_in[14];
  const float* atom_b2  = (const float*)d_in[15];
  const float* bond_w1  = (const float*)d_in[16];
  const float* bond_b1  = (const float*)d_in[17];
  const float* bond_w2  = (const float*)d_in[18];
  const float* bond_b2  = (const float*)d_in[19];
  const float* rbond_w  = (const float*)d_in[20];
  const float* rbond_b  = (const float*)d_in[21];
  const float* wbond_w  = (const float*)d_in[22];
  const float* wbond_b  = (const float*)d_in[23];
  const int* agraph     = (const int*)d_in[24];
  const int* bgraph     = (const int*)d_in[25];
  const int* node2graph = (const int*)d_in[26];
  const int* emask      = (const int*)d_in[27];
  const int* vmask      = (const int*)d_in[28];
  const int* topo_step  = (const int*)d_in[29];
  const int* topo_bid   = (const int*)d_in[30];
  const int* topo_xid   = (const int*)d_in[31];
  const int* topo_label = (const int*)d_in[32];
  const int* atom_step  = (const int*)d_in[33];
  const int* atom_bid   = (const int*)d_in[34];
  const int* atom_xid   = (const int*)d_in[35];
  const int* atom_label = (const int*)d_in[36];
  const int* bond_step  = (const int*)d_in[37];
  const int* bond_bid   = (const int*)d_in[38];
  const int* bond_zid   = (const int*)d_in[39];
  const int* bond_atype = (const int*)d_in[40];
  const int* bond_label = (const int*)d_in[41];
  const int* bond_gs    = (const int*)d_in[42];

  // bump allocator over d_ws
  size_t off = 0;
  auto alloc = [&](size_t bytes) -> void* {
    void* p = (char*)d_ws + off;
    off = (off + bytes + 255) & ~(size_t)255;
    return p;
  };
  float* pre      = (float*)alloc((size_t)E_ * H_ * 4);
  float* h        = (float*)alloc((size_t)E_ * H_ * 4);
  float* nei      = (float*)alloc((size_t)E_ * H_ * 4); // reused as aggN [N_,H_]
  float* nodes    = (float*)alloc((size_t)T_ * N_ * H_ * 4);
  float* gvec     = (float*)alloc((size_t)T_ * B_ * H_ * 4);
  float* topo_hid = (float*)alloc((size_t)PT_ * H_ * 4);
  float* atom_hid = (float*)alloc((size_t)PA_ * H_ * 4);
  float* wb       = (float*)alloc((size_t)PB_ * H_ * 4);
  float* hist     = (float*)alloc((size_t)PB_ * H_ * 4);
  float* cur      = (float*)alloc((size_t)PB_ * H_ * 4);
  float* bond_hid = (float*)alloc((size_t)PB_ * H_ * 4);

  const int ldWh = rup64(DE_);            // 256
  const int ldUh = rup64(H_);             // 256
  const int ldWo = rup64(DE_ + H_);       // 512
  const int ldT1 = rup64(2 * H_ + L_);    // 576
  const int ldB1 = rup64(3 * H_ + L_);    // 832
  const int ldRb = rup64(H_ + V_);        // 320
  const int ldWb = rup64(H_ + NB_);       // 320
  __bf16* Whb  = (__bf16*)alloc((size_t)H_ * ldWh * 2);
  __bf16* Uhb  = (__bf16*)alloc((size_t)H_ * ldUh * 2);
  __bf16* Wob  = (__bf16*)alloc((size_t)H_ * ldWo * 2);
  __bf16* Tw1b = (__bf16*)alloc((size_t)H_ * ldT1 * 2);
  __bf16* Aw1b = (__bf16*)alloc((size_t)H_ * ldT1 * 2);
  __bf16* Bw1b = (__bf16*)alloc((size_t)H_ * ldB1 * 2);
  __bf16* Rwb  = (__bf16*)alloc((size_t)H_ * ldRb * 2);
  __bf16* Wwb  = (__bf16*)alloc((size_t)H_ * ldWb * 2);

  int* agm = (int*)alloc((size_t)N_ * MAXNB_ * 4);
  int* bgm = (int*)alloc((size_t)E_ * MAXNB_ * 4);
  int* tgi = (int*)alloc((size_t)PT_ * 4);
  int* tni = (int*)alloc((size_t)PT_ * 4);
  int* agi = (int*)alloc((size_t)PA_ * 4);
  int* ani = (int*)alloc((size_t)PA_ * 4);
  int* bgi = (int*)alloc((size_t)PB_ * 4);
  int* bni = (int*)alloc((size_t)PB_ * 4);

  const float inv_b = 1.0f / (float)B_;

  zero_f32_kernel<<<grid1(out_size), 256, 0, stream>>>((float*)d_out, out_size);
  zero_f32_kernel<<<grid1((long long)T_ * B_ * H_), 256, 0, stream>>>(gvec, (long long)T_ * B_ * H_);

  // weight convert + transpose to [NOUT][ldW] bf16 (zero-padded)
  cvt_transpose_bf16<<<grid1((long long)H_ * ldWh), 256, 0, stream>>>(W_h, Whb, DE_, H_, ldWh);
  cvt_transpose_bf16<<<grid1((long long)H_ * ldUh), 256, 0, stream>>>(U_h, Uhb, H_, H_, ldUh);
  cvt_transpose_bf16<<<grid1((long long)H_ * ldWo), 256, 0, stream>>>(W_o, Wob, DE_ + H_, H_, ldWo);
  cvt_transpose_bf16<<<grid1((long long)H_ * ldT1), 256, 0, stream>>>(topo_w1, Tw1b, 2 * H_ + L_, H_, ldT1);
  cvt_transpose_bf16<<<grid1((long long)H_ * ldT1), 256, 0, stream>>>(atom_w1, Aw1b, 2 * H_ + L_, H_, ldT1);
  cvt_transpose_bf16<<<grid1((long long)H_ * ldB1), 256, 0, stream>>>(bond_w1, Bw1b, 3 * H_ + L_, H_, ldB1);
  cvt_transpose_bf16<<<grid1((long long)H_ * ldRb), 256, 0, stream>>>(rbond_w, Rwb, H_ + V_, H_, ldRb);
  cvt_transpose_bf16<<<grid1((long long)H_ * ldWb), 256, 0, stream>>>(wbond_w, Wwb, H_ + NB_, H_, ldWb);

  // combined gather indices for heads
  combine_idx_kernel<<<grid1(PT_), 256, 0, stream>>>(topo_step, topo_bid, B_, tgi, PT_);
  combine_idx_kernel<<<grid1(PT_), 256, 0, stream>>>(topo_step, topo_xid, N_, tni, PT_);
  combine_idx_kernel<<<grid1(PA_), 256, 0, stream>>>(atom_step, atom_bid, B_, agi, PA_);
  combine_idx_kernel<<<grid1(PA_), 256, 0, stream>>>(atom_step, atom_xid, N_, ani, PA_);
  combine_idx_kernel<<<grid1(PB_), 256, 0, stream>>>(bond_step, bond_bid, B_, bgi, PB_);
  combine_idx_kernel<<<grid1(PB_), 256, 0, stream>>>(bond_step, bond_zid, N_, bni, PB_);

  // pre = fmess @ W_h
  {
    GemmArgs g{}; g.seg[0] = segf(fmess, nullptr, DE_, DE_); g.nseg = 1;
    g.W = Whb; g.C = pre; g.M = E_; g.K = DE_; g.NOUT = H_; g.ldW = ldWh;
    launch_gemm(g, stream);
  }

  for (int t = 0; t < T_; ++t) {
    const int* em = emask + (long long)t * E_;
    const int* vm = vmask + (long long)t * N_;
    mask_graph_kernel<<<grid1((long long)N_ * MAXNB_), 256, 0, stream>>>(agraph, em, agm, (long long)N_ * MAXNB_);
    mask_graph_kernel<<<grid1((long long)E_ * MAXNB_), 256, 0, stream>>>(bgraph, em, bgm, (long long)E_ * MAXNB_);
    // depth 0 (h starts at zero): h = relu(pre + b_h) * pad
    init_h_kernel<<<grid1((long long)E_ * H_), 256, 0, stream>>>(pre, b_h, h);
    for (int d = 1; d < DEPTH_; ++d) {
      gather_sum6_kernel<<<grid1((long long)E_ * (H_ / 4)), 256, 0, stream>>>(h, bgm, nei, E_);
      GemmArgs g{}; g.seg[0] = segf(nei, nullptr, H_, H_); g.nseg = 1;
      g.W = Uhb; g.bias = b_h; g.addend = pre; g.C = h;
      g.M = E_; g.K = H_; g.NOUT = H_; g.ldW = ldUh; g.relu = 1; g.zero_row0 = 1;
      launch_gemm(g, stream);
    }
    // node = relu(concat(fnode, h[ag].sum) @ W_o + b_o) * vm
    gather_sum6_kernel<<<grid1((long long)N_ * (H_ / 4)), 256, 0, stream>>>(h, agm, nei, N_);
    float* nodes_t = nodes + (size_t)t * N_ * H_;
    {
      GemmArgs g{}; g.seg[0] = segf(fnode, nullptr, DE_, DE_);
      g.seg[1] = segf(nei, nullptr, H_, H_); g.nseg = 2;
      g.W = Wob; g.bias = b_o; g.C = nodes_t;
      g.M = N_; g.K = DE_ + H_; g.NOUT = H_; g.ldW = ldWo; g.relu = 1; g.rowmask = vm;
      launch_gemm(g, stream);
    }
    segsum_kernel<<<grid1((long long)N_ * H_), 256, 0, stream>>>(nodes_t, node2graph, gvec + (size_t)t * B_ * H_);
  }

  // ---- topo head ----
  {
    GemmArgs g{}; g.seg[0] = segf(gvec, tgi, H_, H_);
    g.seg[1] = segf(nodes, tni, H_, H_);
    g.seg[2] = segf(src, topo_bid, L_, L_); g.nseg = 3;
    g.W = Tw1b; g.bias = topo_b1; g.C = topo_hid;
    g.M = PT_; g.K = 2 * H_ + L_; g.NOUT = H_; g.ldW = ldT1; g.relu = 1;
    launch_gemm(g, stream);
  }
  topo_loss_kernel<<<grid1((long long)PT_ * 32), 256, 0, stream>>>(topo_hid, topo_w2, topo_b2, topo_label, (float*)d_out, inv_b);

  // ---- atom head ----
  {
    GemmArgs g{}; g.seg[0] = segf(gvec, agi, H_, H_);
    g.seg[1] = segf(nodes, ani, H_, H_);
    g.seg[2] = segf(src, atom_bid, L_, L_); g.nseg = 3;
    g.W = Aw1b; g.bias = atom_b1; g.C = atom_hid;
    g.M = PA_; g.K = 2 * H_ + L_; g.NOUT = H_; g.ldW = ldT1; g.relu = 1;
    launch_gemm(g, stream);
  }
  atom_loss_kernel<<<dim3(PA_), dim3(64), 0, stream>>>(atom_hid, atom_w2, atom_b2, atom_label, (float*)d_out, inv_b);

  // ---- bond head ----
  { // wb = relu(concat(zn, onehot_NB(label)) @ wbond + b) * (label>0)
    GemmArgs g{}; g.seg[0] = segf(nodes, bni, H_, H_);
    g.seg[1] = segoh(bond_label, NB_); g.nseg = 2;
    g.W = Wwb; g.bias = wbond_b; g.C = wb;
    g.M = PB_; g.K = H_ + NB_; g.NOUT = H_; g.ldW = ldWb; g.relu = 1; g.rowmask = bond_label;
    launch_gemm(g, stream);
  }
  hist_kernel<<<grid1((long long)PB_ * H_), 256, 0, stream>>>(wb, bond_gs, hist);
  { // cur = relu(concat(hist, onehot_V(atom_type)) @ rbond + b)
    GemmArgs g{}; g.seg[0] = segf(hist, nullptr, H_, H_);
    g.seg[1] = segoh(bond_atype, V_); g.nseg = 2;
    g.W = Rwb; g.bias = rbond_b; g.C = cur;
    g.M = PB_; g.K = H_ + V_; g.NOUT = H_; g.ldW = ldRb; g.relu = 1;
    launch_gemm(g, stream);
  }
  { // bond hidden = relu(concat(gvec, cur, zn, src) @ bond_w1 + b1)
    GemmArgs g{}; g.seg[0] = segf(gvec, bgi, H_, H_);
    g.seg[1] = segf(cur, nullptr, H_, H_);
    g.seg[2] = segf(nodes, bni, H_, H_);
    g.seg[3] = segf(src, bond_bid, L_, L_); g.nseg = 4;
    g.W = Bw1b; g.bias = bond_b1; g.C = bond_hid;
    g.M = PB_; g.K = 3 * H_ + L_; g.NOUT = H_; g.ldW = ldB1; g.relu = 1;
    launch_gemm(g, stream);
  }
  bond_loss_kernel<<<grid1(PB_), 256, 0, stream>>>(bond_hid, bond_w2, bond_b2, bond_label, (float*)d_out, inv_b);

  (void)in_sizes; (void)n_in; (void)ws_size;
}
